// GraphAttentionEmbedding_87763361726822
// MI455X (gfx1250) — compile-verified, matrix operand order checked by reading, not run on hardware
//
#include <hip/hip_runtime.h>

// ---------------------------------------------------------------------------
// Problem constants (from the reference)
// ---------------------------------------------------------------------------
constexpr int NV  = 50000;    // nodes
constexpr int NE  = 500000;   // edges
constexpr int FD  = 128;      // feature dim (IN == OUT == ED)
constexpr int TD  = 64;       // time-encoding dim
constexpr int RAW = 172;      // raw message dim
constexpr int MD  = 64;       // projected message dim

typedef __attribute__((ext_vector_type(16))) _Float16 v16h;
typedef __attribute__((ext_vector_type(8)))  float    v8f;

// ---------------------------------------------------------------------------
// Monotonic float <-> uint encoding for atomicMax on signed floats
// ---------------------------------------------------------------------------
__device__ __forceinline__ unsigned enc_f32(float f) {
    unsigned u = __float_as_uint(f);
    return (u & 0x80000000u) ? ~u : (u | 0x80000000u);
}
__device__ __forceinline__ float dec_f32(unsigned u) {
    unsigned b = (u & 0x80000000u) ? (u ^ 0x80000000u) : ~u;
    return __uint_as_float(b);
}

// ---------------------------------------------------------------------------
// Generic GEMM: D[M,Nout] = X[M,K] @ W[K,Nout] (+ bias[Nout])
// f32 memory, f16 WMMA compute with f32 accumulate.
//
// Template params make everything the compiler needs constant:
//   K    = logical depth, KPAD = K rounded up to 32 (KPAD==K -> no guard path)
//   LDX  = row stride of X, LDW = row stride of W
//
// Grid: x = row-tile walkers (grid-stride over M/16 tiles, 4 waves/block),
//       y = column tile. Each wave pre-loads its B fragments (KPAD/32 v16h
//       registers) ONCE and then streams row tiles: float4 A loads + WMMA.
//
// CDNA5 16-bit A 16x32 layout: lane l (m = l&15, kh = (l>>4)*8):
//   a[j] = A[m][ kh + (j&7) + ((j&8)<<1) ]      (halves j = 0..15)
// B 32x16 layout (n = l&15):
//   b[j] = B[ kh + (j&7) + ((j&8)<<1) ][ n ]
// f32 C/D 16x16: c[r] = C[ r + 8*(l>>4) ][ l&15 ]
// ---------------------------------------------------------------------------
template<int K, int KPAD, int LDX, int LDW>
__global__ __launch_bounds__(128)
void wmma_gemm_kernel(const float* __restrict__ X, int M,
                      const float* __restrict__ W,
                      const float* __restrict__ bias,
                      float* __restrict__ D, int ldd)
{
    constexpr int NSTEP = KPAD / 32;
    const int lane  = threadIdx.x & 31;
    const int wave  = threadIdx.x >> 5;
    const int r15   = lane & 15;
    const int kh    = (lane >> 4) << 3;            // 0 or 8
    const int n     = (blockIdx.y << 4) + r15;     // output column
    const int tiles_m = M >> 4;                    // M is a multiple of 16 here
    const int gwave = blockIdx.x * 4 + wave;
    const int nwave = gridDim.x * 4;

    // ---- pre-load B fragments for this column tile (reused for all rows) ----
    v16h bf[NSTEP];
#pragma unroll
    for (int s = 0; s < NSTEP; ++s) {
#pragma unroll
        for (int j = 0; j < 16; ++j) {
            const int k = s * 32 + kh + (j & 7) + ((j & 8) << 1);
            if constexpr (KPAD == K) {
                bf[s][j] = (_Float16)W[k * LDW + n];
            } else {
                const int kc = (k < K) ? k : (K - 1);   // always in-bounds
                const float wv = W[kc * LDW + n];        // unconditional load
                bf[s][j] = (k < K) ? (_Float16)wv : (_Float16)0.0f;  // v_cndmask
            }
        }
    }

    const float bvv = bias ? bias[n] : 0.0f;

    for (int tm = gwave; tm < tiles_m; tm += nwave) {
        const int m = (tm << 4) + r15;
        v8f acc = {};
        if constexpr (KPAD == K) {
            // LDX is a multiple of 4 here (128): aligned float4 streaming loads
            const float4* __restrict__ xr = (const float4*)(X + (size_t)m * LDX);
            __builtin_prefetch(X + (size_t)(m + 16 * nwave) * LDX, 0, 3);
#pragma unroll
            for (int s = 0; s < NSTEP; ++s) {
                const int q0 = (s * 32 + kh) >> 2;
                const float4 f0 = xr[q0];
                const float4 f1 = xr[q0 + 1];
                const float4 f2 = xr[q0 + 4];
                const float4 f3 = xr[q0 + 5];
                v16h a;
                a[0]  = (_Float16)f0.x; a[1]  = (_Float16)f0.y;
                a[2]  = (_Float16)f0.z; a[3]  = (_Float16)f0.w;
                a[4]  = (_Float16)f1.x; a[5]  = (_Float16)f1.y;
                a[6]  = (_Float16)f1.z; a[7]  = (_Float16)f1.w;
                a[8]  = (_Float16)f2.x; a[9]  = (_Float16)f2.y;
                a[10] = (_Float16)f2.z; a[11] = (_Float16)f2.w;
                a[12] = (_Float16)f3.x; a[13] = (_Float16)f3.y;
                a[14] = (_Float16)f3.z; a[15] = (_Float16)f3.w;
                acc = __builtin_amdgcn_wmma_f32_16x16x32_f16(
                          false, a, false, bf[s], (short)0, acc, false, false);
            }
        } else {
            // K-padded path (msg projection, K=172): clamped unconditional
            // scalar loads + branchless zero-select.
            const float* __restrict__ xrow = X + (size_t)m * LDX;
#pragma unroll
            for (int s = 0; s < NSTEP; ++s) {
                v16h a;
#pragma unroll
                for (int j = 0; j < 16; ++j) {
                    const int k  = s * 32 + kh + (j & 7) + ((j & 8) << 1);
                    const int kc = (k < K) ? k : (K - 1);
                    const float xv = xrow[kc];
                    a[j] = (k < K) ? (_Float16)xv : (_Float16)0.0f;
                }
                acc = __builtin_amdgcn_wmma_f32_16x16x32_f16(
                          false, a, false, bf[s], (short)0, acc, false, false);
            }
        }
        const int row0 = (tm << 4) + ((lane >> 4) << 3);
        float* __restrict__ drow = D + (size_t)row0 * ldd + n;
#pragma unroll
        for (int r = 0; r < 8; ++r)
            drow[(size_t)r * ldd] = acc[r] + bvv;
    }
}

template<int K, int KPAD, int LDX, int LDW>
static inline void launch_gemm(const float* X, int M, const float* W,
                               const float* bias, float* D, int ldd, int Nout,
                               hipStream_t s)
{
    int tiles_m = M >> 4;
    int gx = (tiles_m + 3) / 4;
    if (gx > 1024) gx = 1024;                 // grid-stride over row tiles
    dim3 grid(gx, Nout >> 4);
    wmma_gemm_kernel<K, KPAD, LDX, LDW><<<grid, 128, 0, s>>>(X, M, W, bias, D, ldd);
}

// ---------------------------------------------------------------------------
// Edge time encoding: edge_attr[e, 0:64] = cos(rel_t * time_w + time_b)
// ---------------------------------------------------------------------------
__global__ __launch_bounds__(256)
void time_edge_kernel(const float* __restrict__ lu, const int* __restrict__ src,
                      const float* __restrict__ t,
                      const float* __restrict__ tw, const float* __restrict__ tb,
                      float* __restrict__ edge_attr, int E)
{
    int idx = blockIdx.x * blockDim.x + threadIdx.x;
    if (idx >= E * TD) return;
    int e = idx >> 6, d = idx & 63;
    float rt = lu[src[e]] - t[e];
    edge_attr[(size_t)e * FD + d] = __cosf(rt * tw[d] + tb[d]);
}

// ---------------------------------------------------------------------------
// Per-layer init: agg = 0, amax = 0 (encoded "below any finite"), den = 0
// ---------------------------------------------------------------------------
__global__ __launch_bounds__(256)
void init_layer_kernel(float* __restrict__ agg, unsigned* __restrict__ amax,
                       float* __restrict__ den, int n)
{
    int idx = blockIdx.x * blockDim.x + threadIdx.x;
    if (idx < n * FD) agg[idx] = 0.0f;
    if (idx < n * 2) { amax[idx] = 0u; den[idx] = 0.0f; }
}

// ---------------------------------------------------------------------------
// alpha[e,h] = dot(q[dst], k[src]+e) / sqrt(64); segment-max via atomicMax
// One wave per edge; lanes 0-15 -> head 0, lanes 16-31 -> head 1 (4 ch each).
// ---------------------------------------------------------------------------
__global__ __launch_bounds__(256)
void attn_score_kernel(const float* __restrict__ q, const float* __restrict__ k,
                       const float* __restrict__ ep,
                       const int* __restrict__ src, const int* __restrict__ dst,
                       float* __restrict__ alpha, unsigned* __restrict__ amax, int E)
{
    int e = blockIdx.x * 8 + (threadIdx.x >> 5);
    if (e >= E) return;
    int lane = threadIdx.x & 31;
    int si = src[e], di = dst[e];
    const float4 qv = *(const float4*)(q  + (size_t)di * FD + lane * 4);
    const float4 kv = *(const float4*)(k  + (size_t)si * FD + lane * 4);
    const float4 ev = *(const float4*)(ep + (size_t)e  * FD + lane * 4);
    float p = qv.x * (kv.x + ev.x) + qv.y * (kv.y + ev.y)
            + qv.z * (kv.z + ev.z) + qv.w * (kv.w + ev.w);
#pragma unroll
    for (int off = 8; off; off >>= 1) p += __shfl_xor(p, off, 32);
    if ((lane & 15) == 0) {
        int h = lane >> 4;
        float a = p * 0.125f;                         // 1/sqrt(64)
        alpha[(size_t)e * 2 + h] = a;
        atomicMax(&amax[(size_t)di * 2 + h], enc_f32(a));
    }
}

// ---------------------------------------------------------------------------
// ex = exp(alpha - max); den[dst,h] += ex   (alpha overwritten with ex)
// ---------------------------------------------------------------------------
__global__ __launch_bounds__(256)
void attn_exp_kernel(const int* __restrict__ dst, float* __restrict__ alpha,
                     const unsigned* __restrict__ amax, float* __restrict__ den, int E)
{
    int idx = blockIdx.x * blockDim.x + threadIdx.x;
    if (idx >= E * 2) return;
    int e = idx >> 1, h = idx & 1;
    int di = dst[e];
    float ex = __expf(alpha[idx] - dec_f32(amax[(size_t)di * 2 + h]));
    alpha[idx] = ex;
    atomicAdd(&den[(size_t)di * 2 + h], ex);
}

// ---------------------------------------------------------------------------
// agg[dst] += (v[src] + e) * (ex / den[dst]);  one wave per edge
// ---------------------------------------------------------------------------
__global__ __launch_bounds__(256)
void aggregate_kernel(const float* __restrict__ v, const float* __restrict__ ep,
                      const float* __restrict__ alpha, const float* __restrict__ den,
                      const int* __restrict__ src, const int* __restrict__ dst,
                      float* __restrict__ agg, int E)
{
    int e = blockIdx.x * 8 + (threadIdx.x >> 5);
    if (e >= E) return;
    int lane = threadIdx.x & 31;
    int si = src[e], di = dst[e];
    int h = lane >> 4;
    float a = alpha[(size_t)e * 2 + h] / den[(size_t)di * 2 + h];
    float4 vv = *(const float4*)(v  + (size_t)si * FD + lane * 4);
    float4 ev = *(const float4*)(ep + (size_t)e  * FD + lane * 4);
    float* da = agg + (size_t)di * FD + lane * 4;
    atomicAdd(da + 0, (vv.x + ev.x) * a);
    atomicAdd(da + 1, (vv.y + ev.y) * a);
    atomicAdd(da + 2, (vv.z + ev.z) * a);
    atomicAdd(da + 3, (vv.w + ev.w) * a);
}

// ---------------------------------------------------------------------------
// out = [relu](LayerNorm(agg + skip) * g + b);  one wave per node
// ---------------------------------------------------------------------------
__global__ __launch_bounds__(256)
void finalize_kernel(const float* __restrict__ agg, const float* __restrict__ s,
                     const float* __restrict__ g, const float* __restrict__ b,
                     float* __restrict__ out, int n, int relu)
{
    int node = blockIdx.x * 8 + (threadIdx.x >> 5);
    if (node >= n) return;
    int lane = threadIdx.x & 31;
    float4 av = *(const float4*)(agg + (size_t)node * FD + lane * 4);
    float4 sv = *(const float4*)(s   + (size_t)node * FD + lane * 4);
    float h0 = av.x + sv.x, h1 = av.y + sv.y, h2 = av.z + sv.z, h3 = av.w + sv.w;
    float sum = h0 + h1 + h2 + h3;
    float sq  = h0 * h0 + h1 * h1 + h2 * h2 + h3 * h3;
#pragma unroll
    for (int off = 16; off; off >>= 1) {
        sum += __shfl_xor(sum, off, 32);
        sq  += __shfl_xor(sq,  off, 32);
    }
    float mean = sum * (1.0f / 128.0f);
    float var  = sq * (1.0f / 128.0f) - mean * mean;
    float r    = rsqrtf(var + 1e-5f);
    float4 gv = *(const float4*)(g + lane * 4);
    float4 bv = *(const float4*)(b + lane * 4);
    float o0 = (h0 - mean) * r * gv.x + bv.x;
    float o1 = (h1 - mean) * r * gv.y + bv.y;
    float o2 = (h2 - mean) * r * gv.z + bv.z;
    float o3 = (h3 - mean) * r * gv.w + bv.w;
    if (relu) {
        o0 = fmaxf(o0, 0.0f); o1 = fmaxf(o1, 0.0f);
        o2 = fmaxf(o2, 0.0f); o3 = fmaxf(o3, 0.0f);
    }
    *(float4*)(out + (size_t)node * FD + lane * 4) = make_float4(o0, o1, o2, o3);
}

// ---------------------------------------------------------------------------
// Launch
// Input order (setup_inputs dict, depth-first):
//  0:x 1:last_update 2:edge_index 3:t 4:msg 5:time_w 6:time_b 7:msg_w 8:msg_b
//  9+11*l .. : layer l params {Wq,bq,Wk,bk,Wv,bv,We,Ws,bs,ln_g,ln_b}
// Workspace: ~670 MB of f32 scratch.
// ---------------------------------------------------------------------------
extern "C" void kernel_launch(void* const* d_in, const int* in_sizes, int n_in,
                              void* d_out, int out_size, void* d_ws, size_t ws_size,
                              hipStream_t stream)
{
    const float* x      = (const float*)d_in[0];
    const float* lu     = (const float*)d_in[1];
    const int*   ei     = (const int*)  d_in[2];
    const float* tt     = (const float*)d_in[3];
    const float* msg    = (const float*)d_in[4];
    const float* time_w = (const float*)d_in[5];
    const float* time_b = (const float*)d_in[6];
    const float* msg_w  = (const float*)d_in[7];
    const float* msg_b  = (const float*)d_in[8];
    const int* src = ei;
    const int* dst = ei + NE;

    float* ws = (float*)d_ws;
    size_t off = 0;
    float* edge_attr = ws + off; off += (size_t)NE * FD;
    float* eproj     = ws + off; off += (size_t)NE * FD;
    float* qb        = ws + off; off += (size_t)NV * FD;
    float* kb        = ws + off; off += (size_t)NV * FD;
    float* vb        = ws + off; off += (size_t)NV * FD;
    float* sb        = ws + off; off += (size_t)NV * FD;
    float* hbuf      = ws + off; off += (size_t)NV * FD;
    float* agg       = ws + off; off += (size_t)NV * FD;
    float* alpha     = ws + off; off += (size_t)NE * 2;
    unsigned* amax   = (unsigned*)(ws + off); off += (size_t)NV * 2;
    float* den       = ws + off; off += (size_t)NV * 2;

    // --- edge features: [time_enc(64) | msg @ msg_w + msg_b (64)] ----------
    time_edge_kernel<<<(NE * TD + 255) / 256, 256, 0, stream>>>(
        lu, src, tt, time_w, time_b, edge_attr, NE);
    launch_gemm<RAW, 192, RAW, MD>(msg, NE, msg_w, msg_b, edge_attr + TD, FD, MD, stream);

    const float* hin = x;
    for (int l = 0; l < 3; ++l) {
        const float* Wq  = (const float*)d_in[9 + 11 * l + 0];
        const float* bq  = (const float*)d_in[9 + 11 * l + 1];
        const float* Wk  = (const float*)d_in[9 + 11 * l + 2];
        const float* bk_ = (const float*)d_in[9 + 11 * l + 3];
        const float* Wv  = (const float*)d_in[9 + 11 * l + 4];
        const float* bv_ = (const float*)d_in[9 + 11 * l + 5];
        const float* We  = (const float*)d_in[9 + 11 * l + 6];
        const float* Ws  = (const float*)d_in[9 + 11 * l + 7];
        const float* bs  = (const float*)d_in[9 + 11 * l + 8];
        const float* lg  = (const float*)d_in[9 + 11 * l + 9];
        const float* lb  = (const float*)d_in[9 + 11 * l + 10];

        // node projections + edge projection (WMMA GEMMs, K=128, no guards)
        launch_gemm<FD, FD, FD, FD>(hin, NV, Wq, bq,  qb, FD, FD, stream);
        launch_gemm<FD, FD, FD, FD>(hin, NV, Wk, bk_, kb, FD, FD, stream);
        launch_gemm<FD, FD, FD, FD>(hin, NV, Wv, bv_, vb, FD, FD, stream);
        launch_gemm<FD, FD, FD, FD>(hin, NV, Ws, bs,  sb, FD, FD, stream);
        launch_gemm<FD, FD, FD, FD>(edge_attr, NE, We, nullptr, eproj, FD, FD, stream);

        init_layer_kernel<<<(NV * FD + 255) / 256, 256, 0, stream>>>(agg, amax, den, NV);
        attn_score_kernel<<<(NE + 7) / 8, 256, 0, stream>>>(
            qb, kb, eproj, src, dst, alpha, amax, NE);
        attn_exp_kernel<<<(NE * 2 + 255) / 256, 256, 0, stream>>>(
            dst, alpha, amax, den, NE);
        aggregate_kernel<<<(NE + 7) / 8, 256, 0, stream>>>(
            vb, eproj, alpha, den, src, dst, agg, NE);

        float* outp = (l == 2) ? (float*)d_out : hbuf;
        finalize_kernel<<<(NV + 7) / 8, 256, 0, stream>>>(
            agg, sb, lg, lb, outp, NV, (l < 2) ? 1 : 0);
        hin = hbuf;
    }
}
